// RectifiedLinearAttention_70050916598320
// MI455X (gfx1250) — matrix operand order repaired and checked by reading, not compiled, not run on hardware
//
#include <hip/hip_runtime.h>

#define DIMM 512
#define NTOK 1024
#define NBATCH 16
#define NHEADS 8
#define DHEAD 64
#define ATTN_SCALE 0.125f   // 64^-0.5

typedef __attribute__((ext_vector_type(16))) _Float16 v16h;
typedef __attribute__((ext_vector_type(8)))  _Float16 v8h;
typedef __attribute__((ext_vector_type(8)))  float    v8f;
typedef __attribute__((ext_vector_type(4)))  float    v4f;
typedef __attribute__((ext_vector_type(4)))  int      v4i;

// ---- CDNA5 async global->LDS path (guarded; falls back to sync staging) ----
#if defined(__gfx1250__) && __has_builtin(__builtin_amdgcn_global_load_async_to_lds_b128)
#define HAVE_ASYNC_LDS 1
#else
#define HAVE_ASYNC_LDS 0
#endif

#if HAVE_ASYNC_LDS
typedef __attribute__((address_space(1))) v4i* as1_v4i;
typedef __attribute__((address_space(3))) v4i* as3_v4i;
#endif

template <int N>
static __device__ __forceinline__ void wait_async() {
#if HAVE_ASYNC_LDS
#if __has_builtin(__builtin_amdgcn_s_wait_asynccnt)
  __builtin_amdgcn_s_wait_asynccnt(N);
#else
  asm volatile("s_wait_asynccnt %0" ::"n"(N) : "memory");
#endif
#endif
}

// copy 16 f16 (32B) global -> LDS; same INST_OFFSET applies to both addresses.
static __device__ __forceinline__ void cp32h(_Float16* l, const _Float16* g) {
#if HAVE_ASYNC_LDS
  __builtin_amdgcn_global_load_async_to_lds_b128((as1_v4i)g, (as3_v4i)l, 0, 0);
  __builtin_amdgcn_global_load_async_to_lds_b128((as1_v4i)g, (as3_v4i)l, 16, 0);
#else
  *(v8h*)l       = *(const v8h*)g;
  *(v8h*)(l + 8) = *(const v8h*)(g + 8);
#endif
}

static __device__ __forceinline__ v8f wmma_f16(v16h a, v16h b, v8f c) {
  // D = A(16x32 f16) * B(32x16 f16) + C(16x16 f32)
  return __builtin_amdgcn_wmma_f32_16x16x32_f16(false, a, false, b, (short)0, c,
                                                false, false);
}

// A-fragment (16x32 f16) from row-major [M][K], ld in halves (multiple of 8).
// ISA 7.12.2: lanes 0-15 row m0+lane, K=k0+0..7 & k0+16..23; lanes 16-31 row
// m0+lane-16, K=k0+8..15 & k0+24..31. Two aligned 16B loads.
static __device__ __forceinline__ v16h load_frag_a(const _Float16* base, int ld,
                                                   int m0, int k0, int lane) {
  const _Float16* p = base + (m0 + (lane & 15)) * ld + k0 + (lane >> 4) * 8;
  v8h lo = *(const v8h*)p;
  v8h hi = *(const v8h*)(p + 16);
  v16h r;
#pragma unroll
  for (int i = 0; i < 8; ++i) { r[i] = lo[i]; r[i + 8] = hi[i]; }
  return r;
}

// B-fragment (32x16 f16) from transposed [N][K] storage: lanes 0-15 column
// n0+lane with K=k0+0..7 & k0+16..23; lanes 16-31 K=k0+8..15 & k0+24..31.
static __device__ __forceinline__ v16h load_frag_b(const _Float16* base, int ld,
                                                   int n0, int k0, int lane) {
  const _Float16* p = base + (n0 + (lane & 15)) * ld + k0 + (lane >> 4) * 8;
  v8h lo = *(const v8h*)p;
  v8h hi = *(const v8h*)(p + 16);
  v16h r;
#pragma unroll
  for (int i = 0; i < 8; ++i) { r[i] = lo[i]; r[i + 8] = hi[i]; }
  return r;
}

// ---------------------------------------------------------------------------
// Prep: transpose weights to f16 [N][K] so B-fragments are contiguous loads.
// ---------------------------------------------------------------------------
__global__ void __launch_bounds__(256)
prep_weights(const float* __restrict__ Wqkv, const float* __restrict__ Wo,
             _Float16* __restrict__ WqkvT, _Float16* __restrict__ WoT) {
  const int idx = blockIdx.x * 256 + threadIdx.x;
  if (idx < 3 * DIMM * DIMM) {            // 1536 x 512
    const int n = idx >> 9, k = idx & 511;
    WqkvT[idx] = (_Float16)Wqkv[k * (3 * DIMM) + n];
  }
  if (idx < DIMM * DIMM) {                // 512 x 512
    const int n = idx >> 9, k = idx & 511;
    WoT[idx] = (_Float16)Wo[k * DIMM + n];
  }
}

// ---------------------------------------------------------------------------
// Tiled WMMA GEMM: C[M x N] = A[M x 512] * Bt[N x 512]^T.
// Block tile 128x128, 8 waves as 2(M) x 4(N), each wave 64x32 = 8 wmma tiles.
// Double-buffered LDS with async global->LDS staging of the f16 operands.
// MODE 0: A fp32 (x), scatter f16 epilogue -> Q/K [bh][n][64], V^T [bh][64][n].
// MODE 1: A f16 (normalized rows), fp32 epilogue with bias -> d_out.
// ---------------------------------------------------------------------------
template <int MODE>
__global__ void __launch_bounds__(256)
gemm_wmma(const float* __restrict__ Af32, const _Float16* __restrict__ Af16,
          const _Float16* __restrict__ Bt,
          _Float16* __restrict__ qh, _Float16* __restrict__ kh,
          _Float16* __restrict__ vt,
          float* __restrict__ outp, const float* __restrict__ bias) {
  __shared__ _Float16 As[2][128 * 40];
  __shared__ _Float16 Bs[2][128 * 40];
  const int tid = threadIdx.x, lane = tid & 31, wid = tid >> 5;
  const int wrow = wid >> 2, wcol = wid & 3;
  const int m0b = blockIdx.y * 128, n0b = blockIdx.x * 128;

  const v8f vzero = {0.f, 0.f, 0.f, 0.f, 0.f, 0.f, 0.f, 0.f};
  v8f acc[4][2];
#pragma unroll
  for (int i = 0; i < 4; ++i) { acc[i][0] = vzero; acc[i][1] = vzero; }

  const int r = tid >> 1, t = tid & 1;   // 2 threads per tile row, 16 cols each
  auto stage = [&](int buf, int k0) {
    if (MODE == 0) {
      // fp32 -> f16 conversion must transit VGPRs.
      const v4f* s4 = (const v4f*)(Af32 + (m0b + r) * DIMM + k0 + t * 16);
      if (k0 + 32 < DIMM)   // prefetch the tile after next into GL2/L0
        __builtin_prefetch(Af32 + (m0b + r) * DIMM + k0 + 32 + t * 16, 0, 3);
      v4f f0 = s4[0], f1 = s4[1], f2 = s4[2], f3 = s4[3];
      v8h h0, h1;
#pragma unroll
      for (int i = 0; i < 4; ++i) {
        h0[i] = (_Float16)f0[i]; h0[i + 4] = (_Float16)f1[i];
        h1[i] = (_Float16)f2[i]; h1[i + 4] = (_Float16)f3[i];
      }
      *(v8h*)(As[buf] + r * 40 + t * 16)     = h0;
      *(v8h*)(As[buf] + r * 40 + t * 16 + 8) = h1;
    } else {
      cp32h(As[buf] + r * 40 + t * 16, Af16 + (m0b + r) * DIMM + k0 + t * 16);
    }
    cp32h(Bs[buf] + r * 40 + t * 16, Bt + (n0b + r) * DIMM + k0 + t * 16);
  };

  stage(0, 0);
  for (int it = 0; it < 16; ++it) {      // K = 512 in steps of 32
    const int buf = it & 1;
    if (it + 1 < 16) {
      stage(buf ^ 1, (it + 1) * 32);     // next tile in flight during compute
      wait_async<MODE == 0 ? 2 : 4>();   // retire exactly the current tile
    } else {
      wait_async<0>();
    }
    __syncthreads();
    v16h bf0 = load_frag_b(Bs[buf], 40, wcol * 32, 0, lane);
    v16h bf1 = load_frag_b(Bs[buf], 40, wcol * 32 + 16, 0, lane);
#pragma unroll
    for (int mt = 0; mt < 4; ++mt) {
      v16h af = load_frag_a(As[buf], 40, wrow * 64 + mt * 16, 0, lane);
      acc[mt][0] = wmma_f16(af, bf0, acc[mt][0]);
      acc[mt][1] = wmma_f16(af, bf1, acc[mt][1]);
    }
    __syncthreads();                     // tile consumed; safe to overwrite
  }

  // Epilogue. C layout: lane holds column (lane&15); VGPR i holds row i
  // (lanes 0-15) or row 8+i (lanes 16-31).
  const int cn = lane & 15, rb = (lane >> 4) * 8;
#pragma unroll
  for (int mt = 0; mt < 4; ++mt)
#pragma unroll
    for (int nt = 0; nt < 2; ++nt)
#pragma unroll
      for (int i = 0; i < 8; ++i) {
        const int gm = m0b + wrow * 64 + mt * 16 + rb + i;
        const int gn = n0b + wcol * 32 + nt * 16 + cn;
        const float v = acc[mt][nt][i];
        if (MODE == 0) {
          const int bb = gm >> 10, tok = gm & 1023;
          const _Float16 hv = (_Float16)v;
          if (gn < 512) {
            const int h = gn >> 6, d = gn & 63;
            qh[((bb * NHEADS + h) * NTOK + tok) * DHEAD + d] = hv;
          } else if (gn < 1024) {
            const int c = gn - 512, h = c >> 6, d = c & 63;
            kh[((bb * NHEADS + h) * NTOK + tok) * DHEAD + d] = hv;
          } else {
            const int c = gn - 1024, h = c >> 6, d = c & 63;
            vt[((bb * NHEADS + h) * DHEAD + d) * NTOK + tok] = hv;  // V^T
          }
        } else {
          outp[gm * DIMM + gn] = v + bias[gn];
        }
      }
}

// ---------------------------------------------------------------------------
// ReLU attention, streaming (no softmax => no running max/renorm).
// Block = (b,h, 128 query rows); 8 waves, 16 q-rows each. Q fragments stay in
// registers the whole kernel. Key/value tiles of 64 are double-buffered in
// LDS and staged with async copies so the next tile flies during the WMMAs.
// ---------------------------------------------------------------------------
__global__ void __launch_bounds__(256)
attn_kernel(const _Float16* __restrict__ qh, const _Float16* __restrict__ kh,
            const _Float16* __restrict__ vt, float* __restrict__ pre) {
  __shared__ _Float16 Ks[2][64 * 72];     // keys  [j][dh], 2 x 9.0 KB
  __shared__ _Float16 Vs[2][64 * 72];     // V^T   [d][j],  2 x 9.0 KB
  __shared__ _Float16 Ps[8 * 16 * 72];    // per-wave P tiles, 18.0 KB
  const int tid = threadIdx.x, lane = tid & 31, wid = tid >> 5;
  const int bh = blockIdx.y;
  const int q0 = blockIdx.x * 128;

  // Q fragments: 16 rows x 64 (two K=32 fragments), A-layout from global.
  const _Float16* qptr = qh + (bh * NTOK + q0 + wid * 16) * DHEAD;
  v16h qf0 = load_frag_a(qptr, DHEAD, 0, 0, lane);
  v16h qf1 = load_frag_a(qptr, DHEAD, 0, 32, lane);

  const v8f vzero = {0.f, 0.f, 0.f, 0.f, 0.f, 0.f, 0.f, 0.f};
  v8f oacc[4];
#pragma unroll
  for (int i = 0; i < 4; ++i) oacc[i] = vzero;

  _Float16* myP = Ps + wid * 16 * 72;
  const int cn = lane & 15, rb = (lane >> 4) * 8;

  // staging: 64 rows x 64 halves per tile; 4 threads/row, 16 halves each.
  const int sr = tid >> 2, st = tid & 3;
  const _Float16* kbase = kh + (bh * NTOK + sr) * DHEAD + st * 16;
  const _Float16* vbase = vt + (bh * DHEAD + sr) * NTOK + st * 16;
  auto stage = [&](int buf, int j0) {
    cp32h(&Ks[buf][sr * 72 + st * 16], kbase + j0 * DHEAD);
    cp32h(&Vs[buf][sr * 72 + st * 16], vbase + j0);
  };

  stage(0, 0);
  for (int jb = 0; jb < 16; ++jb) {       // 1024 keys in blocks of 64
    const int buf = jb & 1;
    if (jb + 1 < 16) {
      stage(buf ^ 1, (jb + 1) * 64);      // 4 async ops/wave stay in flight
      wait_async<4>();
    } else {
      wait_async<0>();
    }
    __syncthreads();

    // S = Q K^T for 4 key sub-tiles of 16; relu(s)*scale -> f16 into myP.
#pragma unroll
    for (int nj = 0; nj < 4; ++nj) {
      v8f s = vzero;
      s = wmma_f16(qf0, load_frag_b(&Ks[buf][0], 72, nj * 16, 0, lane), s);
      s = wmma_f16(qf1, load_frag_b(&Ks[buf][0], 72, nj * 16, 32, lane), s);
#pragma unroll
      for (int i = 0; i < 8; ++i)
        myP[(rb + i) * 72 + nj * 16 + cn] =
            (_Float16)(__builtin_fmaxf(s[i], 0.f) * ATTN_SCALE);
    }
    // O += P V (same-wave LDS producer/consumer: DS ops are in-order).
#pragma unroll
    for (int kk = 0; kk < 2; ++kk) {
      v16h pa = load_frag_a(myP, 72, 0, kk * 32, lane);
#pragma unroll
      for (int nd = 0; nd < 4; ++nd)
        oacc[nd] = wmma_f16(pa, load_frag_b(&Vs[buf][0], 72, nd * 16,
                                            kk * 32, lane),
                            oacc[nd]);
    }
    __syncthreads();                      // tile consumed; safe to overwrite
  }

  // Write pre-LayerNorm output [b][tok][h*64+d] fp32.
  const int bb = bh >> 3, h = bh & 7;
#pragma unroll
  for (int nd = 0; nd < 4; ++nd)
#pragma unroll
    for (int i = 0; i < 8; ++i) {
      const int tok = q0 + wid * 16 + rb + i;
      const int col = h * DHEAD + nd * 16 + cn;
      pre[(bb * NTOK + tok) * DIMM + col] = oacc[nd][i];
    }
}

// ---------------------------------------------------------------------------
// LayerNorm: one wave per 512-wide row, shuffle reduction, f16 output.
// ---------------------------------------------------------------------------
__global__ void __launch_bounds__(256)
ln_kernel(const float* __restrict__ pre, const float* __restrict__ g,
          const float* __restrict__ bta, _Float16* __restrict__ xn) {
  const int lane = threadIdx.x & 31, wid = threadIdx.x >> 5;
  const int row = blockIdx.x * 8 + wid;
  const float* p = pre + row * DIMM + lane * 16;
  v4f f[4];
#pragma unroll
  for (int i = 0; i < 4; ++i) f[i] = ((const v4f*)p)[i];
  float s = 0.f, s2 = 0.f;
#pragma unroll
  for (int i = 0; i < 4; ++i)
#pragma unroll
    for (int j = 0; j < 4; ++j) { const float x = f[i][j]; s += x; s2 += x * x; }
#pragma unroll
  for (int off = 16; off > 0; off >>= 1) {
    s  += __shfl_xor(s,  off, 32);
    s2 += __shfl_xor(s2, off, 32);
  }
  const float mean = s * (1.f / 512.f);
  const float var  = s2 * (1.f / 512.f) - mean * mean;
  const float rstd = rsqrtf(var + 1e-5f);
  v8h o0, o1;
#pragma unroll
  for (int i = 0; i < 4; ++i)
#pragma unroll
    for (int j = 0; j < 4; ++j) {
      const int c = lane * 16 + i * 4 + j;
      const float y = (f[i][j] - mean) * rstd * g[c] + bta[c];
      if (i < 2) o0[i * 4 + j] = (_Float16)y;
      else       o1[(i - 2) * 4 + j] = (_Float16)y;
    }
  *(v8h*)(xn + row * DIMM + lane * 16)     = o0;
  *(v8h*)(xn + row * DIMM + lane * 16 + 8) = o1;
}

// ---------------------------------------------------------------------------
extern "C" void kernel_launch(void* const* d_in, const int* in_sizes, int n_in,
                              void* d_out, int out_size, void* d_ws,
                              size_t ws_size, hipStream_t stream) {
  const float* x    = (const float*)d_in[0];   // [16,1024,512]
  const float* Wqkv = (const float*)d_in[1];   // [512,1536]
  const float* ln_g = (const float*)d_in[2];   // [512]
  const float* ln_b = (const float*)d_in[3];   // [512]
  const float* Wo   = (const float*)d_in[4];   // [512,512]
  const float* bo   = (const float*)d_in[5];   // [512]
  float* out = (float*)d_out;                  // [16,1024,512]

  char* ws = (char*)d_ws;
  _Float16* Qh    = (_Float16*)(ws);                      // 16 MB  [bh][n][64]
  _Float16* Kh    = (_Float16*)(ws + ((size_t)16 << 20)); // 16 MB  [bh][n][64]
  _Float16* Vt    = (_Float16*)(ws + ((size_t)32 << 20)); // 16 MB  [bh][64][n]
  float*    Pre   = (float*)   (ws + ((size_t)48 << 20)); // 32 MB  pre-LN fp32
  _Float16* WqkvT = (_Float16*)(ws + ((size_t)80 << 20)); // 1.5 MB [1536][512]
  _Float16* WoT   = (_Float16*)(ws + ((size_t)82 << 20)); // 0.5 MB [512][512]
  _Float16* Xn    = (_Float16*)(ws);                      // aliases Qh (done)

  // 1) weight transpose + f16 conversion
  prep_weights<<<(3 * DIMM * DIMM + 255) / 256, 256, 0, stream>>>(Wqkv, Wo,
                                                                  WqkvT, WoT);
  // 2) QKV GEMM: M=16384, N=1536
  gemm_wmma<0><<<dim3(12, 128), 256, 0, stream>>>(x, nullptr, WqkvT,
                                                  Qh, Kh, Vt, nullptr, nullptr);
  // 3) ReLU attention: grid = (n/128, b*h)
  attn_kernel<<<dim3(8, NBATCH * NHEADS), 256, 0, stream>>>(Qh, Kh, Vt, Pre);
  // 4) LayerNorm (8 rows per block)
  ln_kernel<<<(NBATCH * NTOK) / 8, 256, 0, stream>>>(Pre, ln_g, ln_b, Xn);
  // 5) Output projection + bias: M=16384, N=512
  gemm_wmma<1><<<dim3(4, 128), 256, 0, stream>>>(nullptr, Xn, WoT,
                                                 nullptr, nullptr, nullptr,
                                                 out, bo);
}
